// EnhancedHierarchicalBertCRF_5308579578525
// MI455X (gfx1250) — matrix operand order repaired and checked by reading, not compiled, not run on hardware
//
#include <hip/hip_runtime.h>
#include <cmath>

typedef _Float16 half_t;
typedef __attribute__((ext_vector_type(4)))  _Float16 h4;
typedef __attribute__((ext_vector_type(16))) _Float16 v16h;
typedef __attribute__((ext_vector_type(8)))  float    v8f;

#define B_   16
#define S_   512
#define H_   768
#define L_   12
#define NH_  12
#define DH_  64
#define FF_  3072
#define K1_  10
#define K2_  50
#define NTOK (B_ * S_)   // 8192

static __device__ __forceinline__ v8f wmma_f16(v16h a, v16h b, v8f c) {
    return __builtin_amdgcn_wmma_f32_16x16x32_f16(false, a, false, b, (short)0, c, false, false);
}

static __device__ __forceinline__ float fast_gelu(float v) {
    // tanh-gelu with overflow-safe tanh(u) = 1 - 2/(exp(2u)+1)
    float u = 0.79788456080286536f * (v + 0.044715f * v * v * v);
    float e = __expf(2.0f * u);
    return v * (1.0f - 1.0f / (e + 1.0f));
}

// ---------------------------------------------------------------------------
// WMMA GEMM: C[M,N] = act(A[M,K] @ W[K,N] + bias)   (act: 0=none, 1=gelu)
// block = 256 threads (8 waves), tile 64x128; wave (wr,wc) owns a 2x2 grid of
// 16x16 tiles (M-tiles wr,wr+2 ; N-tiles wc,wc+4) -> 4 WMMA per K-step from
// 4 fragment loads. LDS is double-buffered and stored in WMMA fragment
// layout so each operand is one contiguous 32B v16h load per lane.
// ---------------------------------------------------------------------------
__global__ __launch_bounds__(256) void gemm_wmma_kernel(
    const float* __restrict__ A, const float* __restrict__ W,
    const float* __restrict__ bias, float* __restrict__ C,
    int M, int K, int N, int gelu_flag)
{
    __shared__ v16h AsF[2][4][32];   // 2 buf x 4 M-tiles :  8 KB
    __shared__ v16h BsF[2][8][32];   // 2 buf x 8 N-tiles : 16 KB
    half_t* Ah = (half_t*)AsF;
    half_t* Bh = (half_t*)BsF;

    const int tid  = threadIdx.x;
    const int lane = tid & 31;
    const int w    = tid >> 5;
    const int wr   = w >> 2;        // M-tiles wr, wr+2
    const int wc   = w & 3;         // N-tiles wc, wc+4
    const int bm   = blockIdx.y * 64;
    const int bn   = blockIdx.x * 128;

    auto stageA = [&](int k0, int buf) {
        half_t* dst = Ah + buf * (4 * 32 * 16);
        #pragma unroll
        for (int it = 0; it < 2; ++it) {
            int e   = tid + it * 256;
            int row = e >> 3;                    // 0..63
            int kq  = (e & 7) << 2;              // 0,4,..,28
            float4 f = *(const float4*)&A[(size_t)(bm + row) * K + (k0 + kq)];
            int ln = (((kq >> 3) & 1) << 4) + (row & 15);
            int i0 = ((kq >> 4) << 3) + (kq & 7);
            h4 p; p[0] = (half_t)f.x; p[1] = (half_t)f.y;
                  p[2] = (half_t)f.z; p[3] = (half_t)f.w;
            *(h4*)&dst[(((row >> 4) * 32) + ln) * 16 + i0] = p;
        }
    };
    auto stageB = [&](int k0, int buf) {
        half_t* dst = Bh + buf * (8 * 32 * 16);
        #pragma unroll
        for (int it = 0; it < 4; ++it) {
            int e  = tid + it * 256;
            int k  = e >> 5;                     // 0..31
            int cq = (e & 31) << 2;              // 0,4,..,124
            float4 f = *(const float4*)&W[(size_t)(k0 + k) * N + (bn + cq)];
            int nt = cq >> 4;
            int ib = ((nt * 32) + (((k >> 3) & 1) << 4) + (cq & 15)) * 16 + ((k >> 4) << 3) + (k & 7);
            dst[ib     ] = (half_t)f.x;          // consecutive n -> +16 halves
            dst[ib + 16] = (half_t)f.y;
            dst[ib + 32] = (half_t)f.z;
            dst[ib + 48] = (half_t)f.w;
        }
    };

    v8f acc00 = {}, acc01 = {}, acc10 = {}, acc11 = {};
    stageA(0, 0); stageB(0, 0);
    __syncthreads();
    for (int k0 = 0; k0 < K; k0 += 32) {
        int cur = (k0 >> 5) & 1;
        if (k0 + 32 < K) { stageA(k0 + 32, cur ^ 1); stageB(k0 + 32, cur ^ 1); }
        v16h a0 = AsF[cur][wr][lane];
        v16h a1 = AsF[cur][wr + 2][lane];
        v16h b0 = BsF[cur][wc][lane];
        v16h b1 = BsF[cur][wc + 4][lane];
        acc00 = wmma_f16(a0, b0, acc00);
        acc01 = wmma_f16(a0, b1, acc01);
        acc10 = wmma_f16(a1, b0, acc10);
        acc11 = wmma_f16(a1, b1, acc11);
        __syncthreads();
    }

    // C/D layout: lane -> N = lane&15, M base = (lane>=16)*8, VGPR r -> row
    const int nl = lane & 15;
    const int mb = (lane >> 4) * 8;
    #pragma unroll
    for (int r = 0; r < 8; ++r) {
        int row0 = bm + wr * 16 + mb + r;
        int row1 = row0 + 32;
        int col0 = bn + wc * 16 + nl;
        int col1 = col0 + 64;
        float v00 = acc00[r] + bias[col0];
        float v01 = acc01[r] + bias[col1];
        float v10 = acc10[r] + bias[col0];
        float v11 = acc11[r] + bias[col1];
        if (gelu_flag) {
            v00 = fast_gelu(v00); v01 = fast_gelu(v01);
            v10 = fast_gelu(v10); v11 = fast_gelu(v11);
        }
        C[(size_t)row0 * N + col0] = v00;
        C[(size_t)row0 * N + col1] = v01;
        C[(size_t)row1 * N + col0] = v10;
        C[(size_t)row1 * N + col1] = v11;
    }
}

// ---------------------------------------------------------------------------
// Attention: one block per (b, h, 16-row q tile).
// q, softmax probs, and V chunks all staged in WMMA fragment layout.
// ---------------------------------------------------------------------------
__global__ __launch_bounds__(256) void attn_kernel(
    const float* __restrict__ Q, const float* __restrict__ Kb,
    const float* __restrict__ V, const int* __restrict__ amask,
    float* __restrict__ O)
{
    __shared__ float sc[16][S_];    // 32 KB raw scores
    __shared__ v16h  qF[2][32];     //  2 KB q fragments (2 K-chunks of 32)
    __shared__ v16h  aF[16][32];    // 16 KB prob fragments (16 K-chunks)
    __shared__ v16h  vF[2][4][32];  //  8 KB V fragments, double buffered
    half_t* qFh = (half_t*)qF;
    half_t* aFh = (half_t*)aF;
    half_t* vFh = (half_t*)vF;

    const int qt = blockIdx.x;      // 0..31
    const int h  = blockIdx.y;      // 0..11
    const int b  = blockIdx.z;      // 0..15
    const int tid = threadIdx.x, lane = tid & 31, w = tid >> 5;
    const size_t base = ((size_t)b * S_) * H_ + (size_t)h * DH_;

    const int hsel = lane >> 4;
    const int ml   = lane & 15;
    const float scale = 0.125f;     // 1/sqrt(64)

    // ---- stage q tile (16x64) into fragment layout: one float4 per thread --
    {
        int m  = tid >> 4;                   // 0..15
        int dq = (tid & 15) << 2;            // 0,4,..,60
        float4 f = *(const float4*)&Q[base + (size_t)(qt * 16 + m) * H_ + dq];
        int kc = dq >> 5, dd = dq & 31;
        int ln = ((dd >> 4) << 4) + m;
        int ii = dd & 15;
        h4 p; p[0] = (half_t)f.x; p[1] = (half_t)f.y;
              p[2] = (half_t)f.z; p[3] = (half_t)f.w;
        *(h4*)&qFh[((kc * 32) + ln) * 16 + ii] = p;
    }
    __syncthreads();

    // ---- phase 1: scores = q @ k^T * scale + mask ----
    for (int kt = w * 4; kt < w * 4 + 4; ++kt) {
        v8f acc = {};
        #pragma unroll
        for (int kc = 0; kc < 2; ++kc) {
            v16h a = qF[kc][lane];
            v16h bf;
            // B[kd][n] = K[n][kd]: 16 contiguous floats -> 4x float4
            const float4* kp = (const float4*)&Kb[base +
                (size_t)(kt * 16 + ml) * H_ + kc * 32 + (hsel << 4)];
            #pragma unroll
            for (int g = 0; g < 4; ++g) {
                float4 f = kp[g];
                bf[4 * g + 0] = (half_t)f.x; bf[4 * g + 1] = (half_t)f.y;
                bf[4 * g + 2] = (half_t)f.z; bf[4 * g + 3] = (half_t)f.w;
            }
            acc = wmma_f16(a, bf, acc);
        }
        const int mb = hsel * 8;
        const int col = kt * 16 + ml;
        const float madd = amask[b * S_ + col] ? 0.0f : -1e9f;
        #pragma unroll
        for (int r = 0; r < 8; ++r)
            sc[mb + r][col] = acc[r] * scale + madd;
    }
    __syncthreads();

    // ---- phase 2: softmax (wave per 2 rows); emit probs in fragment layout --
    #pragma unroll
    for (int rr = 0; rr < 2; ++rr) {
        int row = w * 2 + rr;
        float mx = -1e30f;
        for (int c = lane; c < S_; c += 32) mx = fmaxf(mx, sc[row][c]);
        for (int off = 16; off > 0; off >>= 1) mx = fmaxf(mx, __shfl_xor(mx, off));
        float sum = 0.f;
        for (int c = lane; c < S_; c += 32) {
            float e = __expf(sc[row][c] - mx);
            sc[row][c] = e;
            sum += e;
        }
        for (int off = 16; off > 0; off >>= 1) sum += __shfl_xor(sum, off);
        float inv = 1.0f / sum;
        for (int c = lane; c < S_; c += 32) {
            float p = sc[row][c] * inv;
            int chunk = c >> 5, cc = c & 31;
            int ln = (((cc >> 3) & 1) << 4) + row;
            int ii = ((cc >> 4) << 3) + (cc & 7);
            aFh[((chunk * 32) + ln) * 16 + ii] = (half_t)p;
        }
    }

    // ---- phase 3: ctx = att @ v, V double-buffered per 32-K chunk ----
    auto stageV = [&](int k0, int buf) {
        half_t* dst = vFh + buf * (4 * 32 * 16);
        #pragma unroll
        for (int it = 0; it < 2; ++it) {
            int e  = tid + it * 256;
            int k  = e >> 4;                    // 0..31
            int cq = (e & 15) << 2;             // 0,4,..,60
            float4 f = *(const float4*)&V[base + (size_t)(k0 + k) * H_ + cq];
            int nt = cq >> 4;
            int ib = ((nt * 32) + (((k >> 3) & 1) << 4) + (cq & 15)) * 16 + ((k >> 4) << 3) + (k & 7);
            dst[ib     ] = (half_t)f.x;
            dst[ib + 16] = (half_t)f.y;
            dst[ib + 32] = (half_t)f.z;
            dst[ib + 48] = (half_t)f.w;
        }
    };

    stageV(0, 0);
    __syncthreads();    // covers aF writes and vF buf0
    v8f acc = {};
    for (int k0 = 0; k0 < S_; k0 += 32) {
        int cur = (k0 >> 5) & 1;
        if (k0 + 32 < S_) stageV(k0 + 32, cur ^ 1);
        if (w < 4) {
            v16h a  = aF[k0 >> 5][lane];
            v16h bf = vF[cur][w][lane];
            acc = wmma_f16(a, bf, acc);
        }
        __syncthreads();
    }
    if (w < 4) {
        const int mb = hsel * 8;
        #pragma unroll
        for (int r = 0; r < 8; ++r)
            O[base + (size_t)(qt * 16 + mb + r) * H_ + w * 16 + ml] = acc[r];
    }
}

// ---------------------------------------------------------------------------
// Embeddings: x[t,h] = word[id] + pos[s] + type[tt]
// ---------------------------------------------------------------------------
__global__ __launch_bounds__(256) void embed_kernel(
    const int* __restrict__ ids, const int* __restrict__ tt,
    const float* __restrict__ we, const float* __restrict__ pe,
    const float* __restrict__ te, float* __restrict__ x)
{
    int t = blockIdx.x;
    int s = t % S_;
    int id = ids[t], ty = tt[t];
    for (int h = threadIdx.x; h < H_; h += 256)
        x[(size_t)t * H_ + h] = we[(size_t)id * H_ + h] + pe[s * H_ + h] + te[ty * H_ + h];
}

// ---------------------------------------------------------------------------
// LayerNorm over H=768, optional residual add. One block per token row.
// ---------------------------------------------------------------------------
__global__ __launch_bounds__(256) void ln_kernel(
    const float* __restrict__ inp, const float* __restrict__ resid,
    const float* __restrict__ sc, const float* __restrict__ bi,
    float* __restrict__ out, int has_resid)
{
    __shared__ float red[256];
    int t = blockIdx.x;
    float vals[3];
    float s = 0.f;
    #pragma unroll
    for (int i = 0; i < 3; ++i) {
        int h = threadIdx.x + i * 256;
        float v = inp[(size_t)t * H_ + h];
        if (has_resid) v += resid[(size_t)t * H_ + h];
        vals[i] = v;
        s += v;
    }
    red[threadIdx.x] = s;
    __syncthreads();
    for (int o = 128; o > 0; o >>= 1) {
        if (threadIdx.x < o) red[threadIdx.x] += red[threadIdx.x + o];
        __syncthreads();
    }
    float mu = red[0] * (1.0f / H_);
    __syncthreads();
    float vs = 0.f;
    #pragma unroll
    for (int i = 0; i < 3; ++i) { float d = vals[i] - mu; vs += d * d; }
    red[threadIdx.x] = vs;
    __syncthreads();
    for (int o = 128; o > 0; o >>= 1) {
        if (threadIdx.x < o) red[threadIdx.x] += red[threadIdx.x + o];
        __syncthreads();
    }
    float rstd = rsqrtf(red[0] * (1.0f / H_) + 1e-12f);
    #pragma unroll
    for (int i = 0; i < 3; ++i) {
        int h = threadIdx.x + i * 256;
        out[(size_t)t * H_ + h] = (vals[i] - mu) * rstd * sc[h] + bi[h];
    }
}

// ---------------------------------------------------------------------------
// Classifier heads (K not WMMA-friendly; tiny cost)
// ---------------------------------------------------------------------------
__global__ void cls1_kernel(const float* __restrict__ seq, const float* __restrict__ W,
                            const float* __restrict__ bias, float* __restrict__ out)
{
    int idx = blockIdx.x * blockDim.x + threadIdx.x;
    if (idx >= NTOK * K1_) return;
    int t = idx / K1_, n = idx % K1_;
    float s = bias[n];
    for (int h = 0; h < H_; ++h)
        s += seq[(size_t)t * H_ + h] * W[h * K1_ + n];
    out[idx] = s;
}

__global__ void cls2_kernel(const float* __restrict__ seq, const int* __restrict__ lab1,
                            const float* __restrict__ W, const float* __restrict__ bias,
                            float* __restrict__ out)
{
    int idx = blockIdx.x * blockDim.x + threadIdx.x;
    if (idx >= NTOK * K2_) return;
    int t = idx / K2_, n = idx % K2_;
    float s = bias[n] + W[(H_ + lab1[t]) * K2_ + n];
    for (int h = 0; h < H_; ++h)
        s += seq[(size_t)t * H_ + h] * W[h * K2_ + n];
    out[idx] = s;
}

// ---------------------------------------------------------------------------
// CRF NLL (torchcrf semantics). One block per batch element, thread per tag.
// ---------------------------------------------------------------------------
__global__ __launch_bounds__(64) void crf_kernel(
    const float* __restrict__ em, const int* __restrict__ tags,
    const int* __restrict__ mask, const float* __restrict__ start,
    const float* __restrict__ endv, const float* __restrict__ trans,
    int K, float* __restrict__ loss)
{
    __shared__ float alpha[64], alnew[64];
    int b = blockIdx.x, j = threadIdx.x;
    const float* emb = em + (size_t)b * S_ * K;
    const int* tg = tags + b * S_;
    const int* mk = mask + b * S_;

    if (j < K) alpha[j] = start[j] + emb[j];
    __syncthreads();
    for (int t = 1; t < S_; ++t) {
        if (j < K) {
            float mx = -1e30f;
            for (int i = 0; i < K; ++i) mx = fmaxf(mx, alpha[i] + trans[i * K + j]);
            float s = 0.f;
            for (int i = 0; i < K; ++i) s += __expf(alpha[i] + trans[i * K + j] - mx);
            float nxt = mx + __logf(s) + emb[t * K + j];
            alnew[j] = mk[t] ? nxt : alpha[j];
        }
        __syncthreads();
        if (j < K) alpha[j] = alnew[j];
        __syncthreads();
    }
    if (j == 0) {
        float mx = -1e30f;
        for (int i = 0; i < K; ++i) mx = fmaxf(mx, alpha[i] + endv[i]);
        float s = 0.f;
        for (int i = 0; i < K; ++i) s += __expf(alpha[i] + endv[i] - mx);
        float den = mx + __logf(s);

        int tp = tg[0];
        float num = start[tp] + emb[tp];
        for (int t = 1; t < S_; ++t) {
            int tc = tg[t];
            if (mk[t]) { num += trans[tp * K + tc] + emb[t * K + tc]; tp = tc; }
        }
        num += endv[tp];
        atomicAdd(loss, -(num - den) * (1.0f / B_));
    }
}

__global__ void zero_kernel(float* p) { p[0] = 0.f; }

// ---------------------------------------------------------------------------
extern "C" void kernel_launch(void* const* d_in, const int* in_sizes, int n_in,
                              void* d_out, int out_size, void* d_ws, size_t ws_size,
                              hipStream_t stream)
{
    const int*   input_ids  = (const int*)d_in[0];
    const int*   token_type = (const int*)d_in[1];
    const int*   amask      = (const int*)d_in[2];
    const int*   lab1       = (const int*)d_in[3];
    const int*   lab2       = (const int*)d_in[4];
    const float* word_emb   = (const float*)d_in[5];
    const float* pos_emb    = (const float*)d_in[6];
    const float* type_emb   = (const float*)d_in[7];
    const float* emb_ln_s   = (const float*)d_in[8];
    const float* emb_ln_b   = (const float*)d_in[9];
    const float* Wq = (const float*)d_in[10]; const float* bq = (const float*)d_in[11];
    const float* Wk = (const float*)d_in[12]; const float* bk = (const float*)d_in[13];
    const float* Wv = (const float*)d_in[14]; const float* bv = (const float*)d_in[15];
    const float* Wo = (const float*)d_in[16]; const float* bo = (const float*)d_in[17];
    const float* ln1_s = (const float*)d_in[18]; const float* ln1_b = (const float*)d_in[19];
    const float* Wi = (const float*)d_in[20]; const float* bi = (const float*)d_in[21];
    const float* Wf = (const float*)d_in[22]; const float* bf = (const float*)d_in[23];
    const float* ln2_s = (const float*)d_in[24]; const float* ln2_b = (const float*)d_in[25];
    const float* cls1_W = (const float*)d_in[26]; const float* cls1_b = (const float*)d_in[27];
    const float* cls2_W = (const float*)d_in[28]; const float* cls2_b = (const float*)d_in[29];
    const float* crf1_start = (const float*)d_in[30];
    const float* crf1_end   = (const float*)d_in[31];
    const float* crf1_trans = (const float*)d_in[32];
    const float* crf2_start = (const float*)d_in[33];
    const float* crf2_end   = (const float*)d_in[34];
    const float* crf2_trans = (const float*)d_in[35];

    float* out_loss = (float*)d_out;
    float* out_l1   = out_loss + 1;
    float* out_l2   = out_l1 + (size_t)NTOK * K1_;

    float* ws = (float*)d_ws;
    size_t o = 0;
    float* x  = ws + o; o += (size_t)NTOK * H_;
    float* qb = ws + o; o += (size_t)NTOK * H_;
    float* kb = ws + o; o += (size_t)NTOK * H_;
    float* vb = ws + o; o += (size_t)NTOK * H_;
    float* cb = ws + o; o += (size_t)NTOK * H_;
    float* fb = ws + o;   // NTOK * FF_

    dim3 blk(256);
    embed_kernel<<<NTOK, blk, 0, stream>>>(input_ids, token_type, word_emb, pos_emb, type_emb, x);
    ln_kernel<<<NTOK, blk, 0, stream>>>(x, nullptr, emb_ln_s, emb_ln_b, x, 0);

    dim3 gH(H_ / 128, NTOK / 64);    //  6 x 128
    dim3 gF(FF_ / 128, NTOK / 64);   // 24 x 128
    dim3 gAtt(S_ / 16, NH_, B_);     // 32 x 12 x 16

    for (int l = 0; l < L_; ++l) {
        const size_t hh = (size_t)l * H_ * H_;
        gemm_wmma_kernel<<<gH, blk, 0, stream>>>(x, Wq + hh, bq + l * H_, qb, NTOK, H_, H_, 0);
        gemm_wmma_kernel<<<gH, blk, 0, stream>>>(x, Wk + hh, bk + l * H_, kb, NTOK, H_, H_, 0);
        gemm_wmma_kernel<<<gH, blk, 0, stream>>>(x, Wv + hh, bv + l * H_, vb, NTOK, H_, H_, 0);
        attn_kernel<<<gAtt, blk, 0, stream>>>(qb, kb, vb, amask, cb);
        gemm_wmma_kernel<<<gH, blk, 0, stream>>>(cb, Wo + hh, bo + l * H_, qb, NTOK, H_, H_, 0);
        ln_kernel<<<NTOK, blk, 0, stream>>>(qb, x, ln1_s + l * H_, ln1_b + l * H_, x, 1);
        gemm_wmma_kernel<<<gF, blk, 0, stream>>>(x, Wi + (size_t)l * H_ * FF_, bi + l * FF_,
                                                 fb, NTOK, H_, FF_, 1);
        gemm_wmma_kernel<<<gH, blk, 0, stream>>>(fb, Wf + (size_t)l * FF_ * H_, bf + l * H_,
                                                 qb, NTOK, FF_, H_, 0);
        ln_kernel<<<NTOK, blk, 0, stream>>>(qb, x, ln2_s + l * H_, ln2_b + l * H_, x, 1);
    }

    cls1_kernel<<<(NTOK * K1_ + 255) / 256, blk, 0, stream>>>(x, cls1_W, cls1_b, out_l1);
    cls2_kernel<<<(NTOK * K2_ + 255) / 256, blk, 0, stream>>>(x, lab1, cls2_W, cls2_b, out_l2);

    zero_kernel<<<1, 1, 0, stream>>>(out_loss);
    crf_kernel<<<B_, 64, 0, stream>>>(out_l1, lab1, amask, crf1_start, crf1_end, crf1_trans, K1_, out_loss);
    crf_kernel<<<B_, 64, 0, stream>>>(out_l2, lab2, amask, crf2_start, crf2_end, crf2_trans, K2_, out_loss);
}
